// ApproximateInnerProductDecoder_79422535237678
// MI455X (gfx1250) — matrix operand order repaired and checked by reading, not compiled, not run on hardware
//
#include <hip/hip_runtime.h>
#include <hip/hip_bf16.h>
#include <math.h>

#define N_NODES 16384
#define D_FEAT  256
#define KNN     16
#define ROWS_PER_WG 16
#define CHUNK   2048
#define NCHUNK  8          // 16384 / 2048
#define NCAND   (NCHUNK * KNN)   // 128 candidates per row
#define NEG_BIG (-1.0e30f)

typedef __attribute__((ext_vector_type(8)))  __bf16        v8bf;
typedef __attribute__((ext_vector_type(16))) __bf16        v16bf;
typedef __attribute__((ext_vector_type(8)))  float         v8f;
typedef __attribute__((ext_vector_type(8)))  unsigned short v8us;

static __device__ __forceinline__ unsigned short f2bf(float f) {
    unsigned int u = __float_as_uint(f);
    // round-to-nearest-even truncation to bf16
    unsigned int r = (u + 0x7FFFu + ((u >> 16) & 1u)) >> 16;
    return (unsigned short)r;
}

// ---------------------------------------------------------------------------
// Kernel 1: row norms + fp32 -> bf16 copy of Z. One wave per row.
// ---------------------------------------------------------------------------
__global__ __launch_bounds__(256) void prep_kernel(const float* __restrict__ z,
                                                   unsigned short* __restrict__ zb,
                                                   float* __restrict__ norms) {
    const int w    = threadIdx.x >> 5;
    const int lane = threadIdx.x & 31;
    const int row  = blockIdx.x * 8 + w;

    const float* zr = z + (size_t)row * D_FEAT + lane * 8;
    float4 f0 = *(const float4*)(zr);
    float4 f1 = *(const float4*)(zr + 4);

    float s = f0.x*f0.x + f0.y*f0.y + f0.z*f0.z + f0.w*f0.w
            + f1.x*f1.x + f1.y*f1.y + f1.z*f1.z + f1.w*f1.w;

    v8us pack;
    pack[0] = f2bf(f0.x); pack[1] = f2bf(f0.y);
    pack[2] = f2bf(f0.z); pack[3] = f2bf(f0.w);
    pack[4] = f2bf(f1.x); pack[5] = f2bf(f1.y);
    pack[6] = f2bf(f1.z); pack[7] = f2bf(f1.w);
    *(v8us*)(zb + (size_t)row * D_FEAT + lane * 8) = pack;

    #pragma unroll
    for (int o = 16; o > 0; o >>= 1) s += __shfl_xor(s, o);
    if (lane == 0) norms[row] = sqrtf(s);
}

// ---------------------------------------------------------------------------
// Kernel 2: fused bf16-WMMA Gram-tile + normalization + exact local top-16.
// Grid: (1024 row-groups, 8 column chunks). 256 threads = 8 waves.
// Wave w handles cols [chunk + w*256, +256) as 16 WMMA tiles of 16 cols.
// LDS: fp32 sims[16][2048] = 128 KB (dynamic).
// ---------------------------------------------------------------------------
__global__ __launch_bounds__(256) void simknn_kernel(const unsigned short* __restrict__ zb_,
                                                     const float* __restrict__ norms,
                                                     float* __restrict__ cvals,
                                                     int*   __restrict__ cidx) {
    extern __shared__ float smem[];   // ROWS_PER_WG * CHUNK floats
    const __bf16* zb = reinterpret_cast<const __bf16*>(zb_);

    const int tid  = threadIdx.x;
    const int w    = tid >> 5;
    const int lane = tid & 31;
    const int m    = lane & 15;   // row/col within 16-tile
    const int hi   = lane >> 4;   // lane half (K-split for A, K-half for B)

    const int g0        = blockIdx.x * ROWS_PER_WG;
    const int chunkbase = blockIdx.y * CHUNK;

    // --- A fragments: 16 rows x 256 K, contiguous K order (matches B):
    // lane (hi,m): elems 0..15 -> K = k0 + hi*16 + j
    v16bf afrag[8];
    {
        const __bf16* za = zb + (size_t)(g0 + m) * D_FEAT;
        #pragma unroll
        for (int kk = 0; kk < 8; ++kk) {
            afrag[kk] = *(const v16bf*)(za + kk * 32 + hi * 16);
        }
    }

    // Row norms for the C-layout rows this lane's accumulator holds
    float rn[8];
    #pragma unroll
    for (int r = 0; r < 8; ++r) rn[r] = norms[g0 + hi * 8 + r];

    // --- GEMM over this wave's 16 column tiles
    for (int t = 0; t < 16; ++t) {
        const int colbase = chunkbase + w * 256 + t * 16;
        const __bf16* zc = zb + (size_t)(colbase + m) * D_FEAT;
        if (t + 1 < 16)  // hint next B tile toward the near cache
            __builtin_prefetch((const void*)(zc + 16 * D_FEAT), 0, 3);

        v8f acc = {0.f, 0.f, 0.f, 0.f, 0.f, 0.f, 0.f, 0.f};
        #pragma unroll
        for (int kk = 0; kk < 8; ++kk) {
            // B (K x 16) contiguous K order (matches A): lane (hi,m) holds col m, K = k0 + hi*16 + 0..15
            v16bf bfrag = *(const v16bf*)(zc + kk * 32 + hi * 16);
            acc = __builtin_amdgcn_wmma_f32_16x16x32_bf16(
                false, afrag[kk], false, bfrag, (short)0, acc, false, false);
        }

        const float ncol = norms[colbase + m];
        const int   lcol = w * 256 + t * 16 + m;
        #pragma unroll
        for (int r = 0; r < 8; ++r) {
            // ranking value: fast v_rcp_f32 instead of IEEE divide
            const float s = acc[r] * __builtin_amdgcn_rcpf(fmaf(rn[r], ncol, 1e-8f));
            smem[(hi * 8 + r) * CHUNK + lcol] = s;
        }
    }
    __syncthreads();

    // --- Exact per-row top-16 of this 2048-col chunk (iterative wave argmax,
    //     float4 LDS scans; strict '>' + lowest-column tie-break)
    for (int rr = 0; rr < 2; ++rr) {
        const int R  = w + rr * 8;
        float* rowp        = smem + R * CHUNK;
        const float4* rowp4 = (const float4*)rowp;
        const int gR = g0 + R;
        const size_t cbase = (size_t)gR * NCAND + (size_t)blockIdx.y * KNN;

        for (int it = 0; it < KNN; ++it) {
            float bv = NEG_BIG;
            int   bc = 0;
            #pragma unroll 4
            for (int j = 0; j < CHUNK / 128; ++j) {
                const int    qi = lane + j * 32;      // float4 index
                const float4 q  = rowp4[qi];
                const int    cb = qi * 4;
                if (q.x > bv) { bv = q.x; bc = cb;     }
                if (q.y > bv) { bv = q.y; bc = cb + 1; }
                if (q.z > bv) { bv = q.z; bc = cb + 2; }
                if (q.w > bv) { bv = q.w; bc = cb + 3; }
            }
            #pragma unroll
            for (int o = 16; o > 0; o >>= 1) {
                const float ov = __shfl_xor(bv, o);
                const int   oc = __shfl_xor(bc, o);
                if (ov > bv || (ov == bv && oc < bc)) { bv = ov; bc = oc; }
            }
            if (((bc >> 2) & 31) == lane) rowp[bc] = NEG_BIG;   // retire winner
            if (lane == 0) {
                cvals[cbase + it] = bv;
                cidx [cbase + it] = chunkbase + bc;
            }
        }
    }
}

// ---------------------------------------------------------------------------
// Kernel 3: merge 8x16 candidates -> global top-16 (descending, index
// tie-break like lax.top_k), then fp32 edge dot + sigmoid. Wave per row.
// ---------------------------------------------------------------------------
__global__ __launch_bounds__(256) void merge_kernel(const float* __restrict__ z,
                                                    const float* __restrict__ cvals,
                                                    const int*   __restrict__ cidx,
                                                    float* __restrict__ out) {
    const int w    = threadIdx.x >> 5;
    const int lane = threadIdx.x & 31;
    const int row  = blockIdx.x * 8 + w;

    const size_t base = (size_t)row * NCAND + lane * 4;
    const float4 vv = *(const float4*)(cvals + base);
    const int4   ii = *(const int4*)(cidx + base);
    float v[4]  = {vv.x, vv.y, vv.z, vv.w};
    int   ci[4] = {ii.x, ii.y, ii.z, ii.w};

    const float* zr = z + (size_t)row * D_FEAT;
    const float4 a0 = *(const float4*)(zr + lane * 8);
    const float4 a1 = *(const float4*)(zr + lane * 8 + 4);

    for (int j = 0; j < KNN; ++j) {
        float bv = NEG_BIG; int slot = 0;
        #pragma unroll
        for (int t = 0; t < 4; ++t)
            if (v[t] > bv) { bv = v[t]; slot = t; }

        float rv = bv; int ri = ci[slot]; int rl = lane;
        #pragma unroll
        for (int o = 16; o > 0; o >>= 1) {
            const float ov = __shfl_xor(rv, o);
            const int   oi = __shfl_xor(ri, o);
            const int   ol = __shfl_xor(rl, o);
            if (ov > rv || (ov == rv && oi < ri)) { rv = ov; ri = oi; rl = ol; }
        }
        if (lane == rl) v[slot] = NEG_BIG;   // winning lane retires its entry

        // fp32 edge logit: z[row] . z[ri]  (matches reference numerics)
        const float* zc = z + (size_t)ri * D_FEAT;
        const float4 b0 = *(const float4*)(zc + lane * 8);
        const float4 b1 = *(const float4*)(zc + lane * 8 + 4);
        float d = a0.x*b0.x + a0.y*b0.y + a0.z*b0.z + a0.w*b0.w
                + a1.x*b1.x + a1.y*b1.y + a1.z*b1.z + a1.w*b1.w;
        #pragma unroll
        for (int o = 16; o > 0; o >>= 1) d += __shfl_xor(d, o);

        if (lane == 0) out[(size_t)row * KNN + j] = 1.0f / (1.0f + expf(-d));
    }
}

// ---------------------------------------------------------------------------
extern "C" void kernel_launch(void* const* d_in, const int* in_sizes, int n_in,
                              void* d_out, int out_size, void* d_ws, size_t ws_size,
                              hipStream_t stream) {
    const float* z = (const float*)d_in[0];
    float* out = (float*)d_out;

    char* ws = (char*)d_ws;
    // layout: zb (bf16 16384x256 = 8 MB) | norms (64 KB) | cvals (8 MB) | cidx (8 MB)
    unsigned short* zb    = (unsigned short*)(ws);
    float*          norms = (float*)(ws + 8388608);
    float*          cvals = (float*)(ws + 8454144);
    int*            cidxp = (int*)  (ws + 16842752);

    prep_kernel<<<N_NODES / 8, 256, 0, stream>>>(z, zb, norms);

    const size_t lds_bytes = (size_t)ROWS_PER_WG * CHUNK * sizeof(float); // 128 KB
    simknn_kernel<<<dim3(N_NODES / ROWS_PER_WG, NCHUNK, 1), 256, lds_bytes, stream>>>(
        zb, norms, cvals, cidxp);

    merge_kernel<<<N_NODES / 8, 256, 0, stream>>>(z, cvals, cidxp, out);
}
